// SwinTransformer_51187420234289
// MI455X (gfx1250) — matrix operand order repaired and checked by reading, not compiled, not run on hardware
//
#include <hip/hip_runtime.h>

typedef _Float16 half_t;
typedef __attribute__((ext_vector_type(16))) _Float16 v16h;
typedef __attribute__((ext_vector_type(8)))  _Float16 v8h;
typedef __attribute__((ext_vector_type(8)))  float    v8f;

// gfx1250 async global->LDS load (16B per lane, GV mode), ASYNCcnt-tracked.
__device__ __forceinline__ void async_lds_b128(unsigned lds_off, const void* gaddr) {
  asm volatile("global_load_async_to_lds_b128 %0, %1, off"
               :: "v"(lds_off), "v"(gaddr) : "memory");
}
__device__ __forceinline__ void wait_asynccnt0() {
  asm volatile("s_wait_asynccnt 0x0" ::: "memory");
}

// ---------------------------------------------------------------------------
// Generic WMMA GEMM: C[M,N] = epi(A_f16[M,K] @ B_f16[K,N] + bias) (+res)
// 256 threads = 8 waves; macro tile 128(M) x 64(N); K step 32.
// Wave grid 4x2, each wave owns a 32x32 register tile = 2x2 WMMA tiles.
// A tile is staged via GLOBAL_LOAD_ASYNC_TO_LDS_B128 (fast path) so the
// vector pipe overlaps the B transpose while the DMA fills LDS; completion
// via s_wait_asynccnt. B is stored transposed so fragment reads are b128.
// EPI: 0=none 1=exact GELU.  OUT16: write f16 instead of f32.  RES: += res[m,n].
// ---------------------------------------------------------------------------
template<int EPI, int OUT16, int RES>
__global__ __launch_bounds__(256)
void k_gemm(const half_t* __restrict__ A, const half_t* __restrict__ Bm,
            const float* __restrict__ bias, const float* __restrict__ res,
            void* __restrict__ Cout, int M, int N, int K) {
  __shared__ __align__(16) half_t As[128][32];
  __shared__ __align__(16) half_t BsT[64][32];   // BsT[n][k]
  const int tid  = threadIdx.x;
  const int lane = tid & 31;
  const int wv   = tid >> 5;
  const int wm   = wv >> 1;          // 0..3  (32-row band)
  const int wn   = wv & 1;           // 0..1  (32-col band)
  const int m0   = blockIdx.y * 128;
  const int n0   = blockIdx.x * 64;
  const int l15  = lane & 15;
  const int hid  = lane >> 4;

  v8f acc00 = {}, acc01 = {}, acc10 = {}, acc11 = {};

  for (int k0 = 0; k0 < K; k0 += 32) {
    // ---- A tile: 128x32 halves = 512 x 16B ----
    const bool fullA = (m0 + 128 <= M) && (k0 + 32 <= K);   // uniform
    if (fullA) {
      // async DMA into LDS; overlaps with B transpose below
#pragma unroll
      for (int v = tid; v < 512; v += 256) {
        int r = v >> 2, c = (v & 3) << 3;
        unsigned lo = (unsigned)(size_t)(&As[r][c]);        // LDS byte offset
        async_lds_b128(lo, A + (size_t)(m0 + r) * K + (k0 + c));
      }
    } else {
      for (int v = tid; v < 512; v += 256) {
        int r = v >> 2, c = (v & 3) << 3;
        int gm = m0 + r, gk = k0 + c;
        v8h val = {};
        if (gm < M) {
          if (gk + 7 < K) {
            val = *(const v8h*)(A + (size_t)gm * K + gk);
          } else {
#pragma unroll
            for (int e = 0; e < 8; ++e)
              val[e] = (gk + e < K) ? A[(size_t)gm * K + gk + e] : (half_t)0.f;
          }
        }
        *(v8h*)(&As[r][c]) = val;
      }
    }
    // ---- B tile: 32x64 -> BsT[64][32], 256 x v8h ----
    {
      int v = tid;
      int r = v >> 3, c = (v & 7) << 3;
      int gk = k0 + r, gn = n0 + c;
      v8h val = {};
      if (gk < K) {
        if (gn + 7 < N) {
          val = *(const v8h*)(Bm + (size_t)gk * N + gn);
        } else {
#pragma unroll
          for (int e = 0; e < 8; ++e)
            val[e] = (gn + e < N) ? Bm[(size_t)gk * N + gn + e] : (half_t)0.f;
        }
      }
#pragma unroll
      for (int e = 0; e < 8; ++e) BsT[c + e][r] = val[e];
    }
    // prefetch next K tile into cache while we compute (global_prefetch_b8)
    if (k0 + 32 < K) {
      int pr = m0 + (tid >> 1);
      if (pr < M) __builtin_prefetch(A + (size_t)pr * K + k0 + 32, 0, 1);
      __builtin_prefetch(Bm + (size_t)(k0 + 32 + (tid & 31)) * N + n0, 0, 1);
    }
    if (fullA) wait_asynccnt0();
    __syncthreads();

    v16h af0, af1, bf0, bf1;
    const int arow = wm * 32 + l15;
#pragma unroll
    for (int i = 0; i < 16; ++i) {
      int kk = (hid << 4) + i;                           // 16 contiguous halves
      af0[i] = As[arow][kk];
      af1[i] = As[arow + 16][kk];
    }
    const int bcol = wn * 32 + l15;
#pragma unroll
    for (int i = 0; i < 16; ++i) {
      int kk = (hid << 4) + i;                           // 16 contiguous halves
      bf0[i] = BsT[bcol][kk];
      bf1[i] = BsT[bcol + 16][kk];
    }
    acc00 = __builtin_amdgcn_wmma_f32_16x16x32_f16(false, af0, false, bf0,
                                                   (short)0, acc00, false, false);
    acc01 = __builtin_amdgcn_wmma_f32_16x16x32_f16(false, af0, false, bf1,
                                                   (short)0, acc01, false, false);
    acc10 = __builtin_amdgcn_wmma_f32_16x16x32_f16(false, af1, false, bf0,
                                                   (short)0, acc10, false, false);
    acc11 = __builtin_amdgcn_wmma_f32_16x16x32_f16(false, af1, false, bf1,
                                                   (short)0, acc11, false, false);
    __syncthreads();
  }

  auto epi = [&](const v8f& a, int mt, int nt) {
    const int gn = n0 + wn * 32 + nt * 16 + l15;
#pragma unroll
    for (int r = 0; r < 8; ++r) {
      int gm = m0 + wm * 32 + mt * 16 + r + hid * 8;
      if (gm < M && gn < N) {
        float v = a[r];
        if (bias) v += bias[gn];
        if (EPI == 1) v = 0.5f * v * (1.f + erff(v * 0.70710678118654752f));
        if (RES) v += res[(size_t)gm * N + gn];
        if (OUT16) ((half_t*)Cout)[(size_t)gm * N + gn] = (half_t)v;
        else       ((float*)Cout)[(size_t)gm * N + gn] = v;
      }
    }
  };
  epi(acc00, 0, 0); epi(acc01, 0, 1); epi(acc10, 1, 0); epi(acc11, 1, 1);
}

// ---------------------------------------------------------------------------
// Fused windowed attention: one workgroup (4 waves) per (window, head).
// N = 49 tokens padded to 64, head_dim = 32 always.
// S = (Q*scale) Kt via WMMA -> LDS fp32; +rel-pos-bias (+shift mask, computed
// inline); two-pass softmax in LDS; P (f16, LDS) @ V via WMMA -> out f16.
// V is stored transposed (VT[d][k]) so PV B-fragment reads are contiguous.
// ---------------------------------------------------------------------------
__global__ __launch_bounds__(128)
void k_attn(const half_t* __restrict__ qkv, const float* __restrict__ rpb,
            half_t* __restrict__ out, int nW_side, int nh, int C,
            int H, int shift) {
  __shared__ __align__(16) half_t Qs[64][32];
  __shared__ __align__(16) half_t Ks[64][32];
  __shared__ __align__(16) half_t VT[32][64];
  __shared__ float  Sm[64][64];
  __shared__ __align__(16) half_t Ps[64][64];

  const int w = blockIdx.x;                 // global window index (b*nW + win)
  const int h = blockIdx.y;                 // head
  const int winPerImg = nW_side * nW_side;
  const int win = w % winPerImg;
  const int wh = win / nW_side, ww = win % nW_side;
  const int tid = threadIdx.x;
  const int C3 = 3 * C;
  const float scale = 0.17677669529663687f; // 32^-0.5

  // vectorized Q/K/V load: 64x32 halves = 256 x v8h, 2 per thread
  for (int v = tid; v < 256; v += 128) {
    int n = v >> 2, d = (v & 3) << 3;
    v8h q = {}, kk = {}, vv = {};
    if (n < 49) {
      size_t base = ((size_t)w * 49 + n) * C3 + (size_t)h * 32 + d;
      q  = *(const v8h*)(qkv + base);
      kk = *(const v8h*)(qkv + base + C);
      vv = *(const v8h*)(qkv + base + 2 * C);
#pragma unroll
      for (int e = 0; e < 8; ++e) q[e] = (half_t)((float)q[e] * scale);
    }
    *(v8h*)(&Qs[n][d]) = q;
    *(v8h*)(&Ks[n][d]) = kk;
#pragma unroll
    for (int e = 0; e < 8; ++e) VT[d + e][n] = vv[e];
  }
  __syncthreads();

  const int lane = tid & 31, wv = tid >> 5;
  const int l15 = lane & 15, hid = lane >> 4;
  const int m0 = wv * 16;

  // ---- S = Q @ K^T  (K=32, one WMMA step; 4 N tiles) ----
  v16h af;
#pragma unroll
  for (int i = 0; i < 16; ++i) {
    int kk = (hid << 4) + i;
    af[i] = Qs[m0 + l15][kk];
  }
#pragma unroll
  for (int nt = 0; nt < 4; ++nt) {
    v16h bf;
#pragma unroll
    for (int i = 0; i < 16; ++i)
      bf[i] = Ks[nt * 16 + l15][(hid << 4) + i];     // B[k][n] = K[n][k]
    v8f acc = {};
    acc = __builtin_amdgcn_wmma_f32_16x16x32_f16(false, af, false, bf,
                                                 (short)0, acc, false, false);
#pragma unroll
    for (int r = 0; r < 8; ++r)
      Sm[m0 + r + hid * 8][nt * 16 + l15] = acc[r];
  }
  __syncthreads();

  // ---- softmax rows: bias + inline shifted-window mask, two-pass in LDS ----
  if (tid < 64) {
    const int n = tid;
    if (n < 49) {
      const int r1 = n / 7, c1 = n % 7;
      int reg1 = 0;
      if (shift > 0) {
        int gr = wh * 7 + r1, gc = ww * 7 + c1;
        int idr = (gr < H - 7) ? 0 : ((gr < H - shift) ? 1 : 2);
        int idc = (gc < H - 7) ? 0 : ((gc < H - shift) ? 1 : 2);
        reg1 = idr * 3 + idc;
      }
      float mx = -1e30f;
      for (int m = 0; m < 49; ++m) {
        const int r2 = m / 7, c2 = m % 7;
        float v = Sm[n][m] +
                  rpb[(size_t)((r1 - r2 + 6) * 13 + (c1 - c2 + 6)) * nh + h];
        if (shift > 0) {
          int gr = wh * 7 + r2, gc = ww * 7 + c2;
          int idr = (gr < H - 7) ? 0 : ((gr < H - shift) ? 1 : 2);
          int idc = (gc < H - 7) ? 0 : ((gc < H - shift) ? 1 : 2);
          if (idr * 3 + idc != reg1) v -= 100.f;
        }
        Sm[n][m] = v;
        mx = fmaxf(mx, v);
      }
      float sum = 0.f;
      for (int m = 0; m < 49; ++m) {
        float e = __expf(Sm[n][m] - mx);
        Sm[n][m] = e;
        sum += e;
      }
      float inv = 1.f / sum;
      for (int m = 0; m < 49; ++m) Ps[n][m] = (half_t)(Sm[n][m] * inv);
      for (int m = 49; m < 64; ++m) Ps[n][m] = (half_t)0.f;
    } else {
      for (int m = 0; m < 64; ++m) Ps[n][m] = (half_t)0.f;
    }
  }
  __syncthreads();

  // ---- O = P @ V  (K=64 -> 2 WMMA steps; 2 N tiles of 16) ----
#pragma unroll
  for (int nt = 0; nt < 2; ++nt) {
    v8f acc = {};
#pragma unroll
    for (int ks = 0; ks < 2; ++ks) {
      const int k0 = ks * 32;
      v16h pa, vb;
#pragma unroll
      for (int i = 0; i < 16; ++i) {
        int kk = (hid << 4) + i;
        pa[i] = Ps[m0 + l15][k0 + kk];
      }
#pragma unroll
      for (int i = 0; i < 16; ++i)
        vb[i] = VT[nt * 16 + l15][k0 + (hid << 4) + i];  // contiguous
      acc = __builtin_amdgcn_wmma_f32_16x16x32_f16(false, pa, false, vb,
                                                   (short)0, acc, false, false);
    }
#pragma unroll
    for (int r = 0; r < 8; ++r) {
      int n = m0 + r + hid * 8;
      if (n < 49)
        out[((size_t)w * 49 + n) * C + (size_t)h * 32 + nt * 16 + l15] =
            (half_t)acc[r];
    }
  }
}

// ---------------------------------------------------------------------------
// Support kernels
// ---------------------------------------------------------------------------
__global__ void k_cvt(const float* __restrict__ s, half_t* __restrict__ d, int n) {
  int i = blockIdx.x * blockDim.x + threadIdx.x;
  if (i < n) d[i] = (half_t)s[i];
}

// dst[k][n] = src[n][k]  (patch-embed weight OI(HW) -> [K,N])
__global__ void k_cvt_tr(const float* __restrict__ s, half_t* __restrict__ d,
                         int N, int K) {
  int i = blockIdx.x * blockDim.x + threadIdx.x;
  if (i >= N * K) return;
  int n = i / K, k = i % K;
  d[(size_t)k * N + n] = (half_t)s[i];
}

// im2col for 4x4/stride4 patch embed: A[b*3136 + ph*56 + pw][ci*16+kh*4+kw]
__global__ void k_im2col(const float* __restrict__ img, half_t* __restrict__ out,
                         int Bn) {
  int i = blockIdx.x * blockDim.x + threadIdx.x;
  int total = Bn * 3136 * 48;
  if (i >= total) return;
  int kidx = i % 48, token = i / 48;
  int pw = token % 56, ph = (token / 56) % 56, b = token / 3136;
  int ci = kidx / 16, kh = (kidx >> 2) & 3, kw = kidx & 3;
  out[i] = (half_t)img[((size_t)b * 3 + ci) * 224 * 224 +
                       (size_t)(ph * 4 + kh) * 224 + (pw * 4 + kw)];
}

// LayerNorm, thread per token. OUT16: f16 out (feeds a GEMM) else f32.
template<int OUT16>
__global__ void k_ln(const float* __restrict__ x, const float* __restrict__ g,
                     const float* __restrict__ b, void* __restrict__ out,
                     int M, int C) {
  int t = blockIdx.x * blockDim.x + threadIdx.x;
  if (t >= M) return;
  const float* row = x + (size_t)t * C;
  float mean = 0.f;
  for (int c = 0; c < C; ++c) mean += row[c];
  mean /= C;
  float var = 0.f;
  for (int c = 0; c < C; ++c) { float d = row[c] - mean; var += d * d; }
  var /= C;
  float inv = rsqrtf(var + 1e-5f);
  for (int c = 0; c < C; ++c) {
    float v = (row[c] - mean) * inv * g[c] + b[c];
    if (OUT16) ((half_t*)out)[(size_t)t * C + c] = (half_t)v;
    else       ((float*)out)[(size_t)t * C + c] = v;
  }
}

// LN + cyclic shift + window partition -> f16 [nWB*49, C]
__global__ void k_ln_window(const float* __restrict__ x, const float* __restrict__ g,
                            const float* __restrict__ b, half_t* __restrict__ out,
                            int Bn, int H, int C, int shift) {
  int nws = H / 7;
  int total = Bn * nws * nws * 49;
  int t = blockIdx.x * blockDim.x + threadIdx.x;
  if (t >= total) return;
  int n = t % 49, w = t / 49;
  int win = w % (nws * nws), bimg = w / (nws * nws);
  int wh = win / nws, ww = win % nws;
  int r = n / 7, c = n % 7;
  int sr = (wh * 7 + r + shift) % H;     // roll by -shift
  int sc = (ww * 7 + c + shift) % H;
  const float* row = x + ((size_t)bimg * H * H + (size_t)sr * H + sc) * C;
  float mean = 0.f;
  for (int cc = 0; cc < C; ++cc) mean += row[cc];
  mean /= C;
  float var = 0.f;
  for (int cc = 0; cc < C; ++cc) { float d = row[cc] - mean; var += d * d; }
  var /= C;
  float inv = rsqrtf(var + 1e-5f);
  for (int cc = 0; cc < C; ++cc)
    out[(size_t)t * C + cc] = (half_t)((row[cc] - mean) * inv * g[cc] + b[cc]);
}

// window reverse (+roll back) + residual add into x (fp32, in place)
__global__ void k_wrev_add(float* __restrict__ x, const float* __restrict__ po,
                           int Bn, int H, int C, int shift) {
  size_t total = (size_t)Bn * H * H * C;
  size_t idx = (size_t)blockIdx.x * blockDim.x + threadIdx.x;
  if (idx >= total) return;
  int c = idx % C;
  size_t rest = idx / C;
  int j = rest % H; rest /= H;
  int i = rest % H;
  int si = (i - shift + H) % H;          // roll back by +shift
  int sj = (j - shift + H) % H;
  int nws = H / 7;
  int wh = si / 7, r = si % 7, ww = sj / 7, cc = sj % 7;
  size_t bimg = rest / H;
  size_t w = bimg * nws * nws + (size_t)wh * nws + ww;
  x[idx] += po[(w * 49 + (size_t)(r * 7 + cc)) * C + c];
}

// patch-merging gather + LN over 4C -> f16 [Bn*H2*H2, 4C]
__global__ void k_merge_ln(const float* __restrict__ x, const float* __restrict__ g,
                           const float* __restrict__ b, half_t* __restrict__ out,
                           int Bn, int H, int C) {
  int H2 = H / 2;
  int total = Bn * H2 * H2;
  int t = blockIdx.x * blockDim.x + threadIdx.x;
  if (t >= total) return;
  int j = t % H2, i = (t / H2) % H2, bimg = t / (H2 * H2);
  const float* base = x + (size_t)bimg * H * H * C;
  size_t off[4] = { ((size_t)(2 * i) * H + 2 * j) * C,
                    ((size_t)(2 * i + 1) * H + 2 * j) * C,
                    ((size_t)(2 * i) * H + 2 * j + 1) * C,
                    ((size_t)(2 * i + 1) * H + 2 * j + 1) * C };
  int C4 = 4 * C;
  float mean = 0.f;
  for (int q = 0; q < 4; ++q)
    for (int c = 0; c < C; ++c) mean += base[off[q] + c];
  mean /= C4;
  float var = 0.f;
  for (int q = 0; q < 4; ++q)
    for (int c = 0; c < C; ++c) { float d = base[off[q] + c] - mean; var += d * d; }
  var /= C4;
  float inv = rsqrtf(var + 1e-5f);
  for (int q = 0; q < 4; ++q)
    for (int c = 0; c < C; ++c) {
      int oc = q * C + c;
      out[(size_t)t * C4 + oc] =
          (half_t)((base[off[q] + c] - mean) * inv * g[oc] + b[oc]);
    }
}

// ---------------------------------------------------------------------------
// Host orchestration
// ---------------------------------------------------------------------------
static inline int cdiv(int a, int b) { return (a + b - 1) / b; }

extern "C" void kernel_launch(void* const* d_in, const int* in_sizes, int n_in,
                              void* d_out, int out_size, void* d_ws, size_t ws_size,
                              hipStream_t stream) {
  (void)in_sizes; (void)n_in; (void)out_size; (void)ws_size;
  const int DEPTHS[4] = {2, 2, 6, 2};
  const int HEADS_[4] = {3, 6, 12, 24};
  const int Bn = 16;

  int idx = 0;
  const float* X_IN  = (const float*)d_in[idx++];
  const float* PE_W  = (const float*)d_in[idx++];
  const float* PE_B  = (const float*)d_in[idx++];
  const float* PE_NG = (const float*)d_in[idx++];
  const float* PE_NB = (const float*)d_in[idx++];

  struct BlockP {
    const float *n1g, *n1b, *qkv_w, *qkv_b, *rpb, *proj_w, *proj_b,
                *n2g, *n2b, *fc1_w, *fc1_b, *fc2_w, *fc2_b;
  };
  BlockP blk[4][6];
  const float* ds_ng[4] = {nullptr, nullptr, nullptr, nullptr};
  const float* ds_nb[4] = {nullptr, nullptr, nullptr, nullptr};
  const float* ds_rw[4] = {nullptr, nullptr, nullptr, nullptr};
  for (int s = 0; s < 4; ++s) {
    for (int i = 0; i < DEPTHS[s]; ++i) {
      BlockP& p = blk[s][i];
      p.n1g = (const float*)d_in[idx++]; p.n1b = (const float*)d_in[idx++];
      p.qkv_w = (const float*)d_in[idx++]; p.qkv_b = (const float*)d_in[idx++];
      p.rpb = (const float*)d_in[idx++];
      p.proj_w = (const float*)d_in[idx++]; p.proj_b = (const float*)d_in[idx++];
      p.n2g = (const float*)d_in[idx++]; p.n2b = (const float*)d_in[idx++];
      p.fc1_w = (const float*)d_in[idx++]; p.fc1_b = (const float*)d_in[idx++];
      p.fc2_w = (const float*)d_in[idx++]; p.fc2_b = (const float*)d_in[idx++];
    }
    if (s < 3) {
      ds_ng[s] = (const float*)d_in[idx++];
      ds_nb[s] = (const float*)d_in[idx++];
      ds_rw[s] = (const float*)d_in[idx++];
    }
  }

  // workspace carve (sizes based on stage-0 maxima)
  const size_t TOK0 = (size_t)Bn * 56 * 56;   // 50176
  char* wsb = (char*)d_ws;
  auto carve = [&](size_t bytes) {
    char* p = wsb;
    wsb += (bytes + 255) & ~(size_t)255;
    return p;
  };
  float*  xbuf  = (float*) carve(TOK0 * 96 * 4);        // current activations fp32
  half_t* a16   = (half_t*)carve(TOK0 * 96 * 2);        // LN/partition/merge out, GEMM A
  half_t* qkv16 = (half_t*)carve(TOK0 * 288 * 2);       // qkv GEMM out f16
  half_t* att16 = (half_t*)carve(TOK0 * 96 * 2);        // attention out f16
  half_t* h16   = (half_t*)carve(TOK0 * 384 * 2);       // fc1 out f16
  float*  tmp32 = (float*) carve(TOK0 * 96 * 4);        // proj / patch-embed GEMM out
  half_t* w16   = (half_t*)carve((size_t)768 * 3072 * 2); // weight conversion arena

  const int TPB = 256;

  // ---- patch embed: im2col -> GEMM(K=48,N=96)+bias -> LN -> x ----
  {
    int M = (int)TOK0;
    k_im2col<<<cdiv(M * 48, TPB), TPB, 0, stream>>>(X_IN, a16, Bn);
    k_cvt_tr<<<cdiv(96 * 48, TPB), TPB, 0, stream>>>(PE_W, w16, 96, 48);
    k_gemm<0, 0, 0><<<dim3(cdiv(96, 64), cdiv(M, 128)), 256, 0, stream>>>(
        a16, w16, PE_B, nullptr, tmp32, M, 96, 48);
    k_ln<0><<<cdiv(M, TPB), TPB, 0, stream>>>(tmp32, PE_NG, PE_NB, xbuf, M, 96);
  }

  int H = 56, C = 96;
  size_t out_off = 0;
  for (int s = 0; s < 4; ++s) {
    const int nh = HEADS_[s];
    const int tokens = Bn * H * H;
    const int nws = H / 7;
    const int nWB = Bn * nws * nws;
    for (int i = 0; i < DEPTHS[s]; ++i) {
      const BlockP& p = blk[s][i];
      const int shift = (i % 2 == 1 && H > 7) ? 3 : 0;

      // LN1 + shift + window partition -> a16 [tokens, C] f16
      k_ln_window<<<cdiv(tokens, TPB), TPB, 0, stream>>>(
          xbuf, p.n1g, p.n1b, a16, Bn, H, C, shift);

      // qkv GEMM -> qkv16 f16
      k_cvt<<<cdiv(C * 3 * C, TPB), TPB, 0, stream>>>(p.qkv_w, w16, C * 3 * C);
      k_gemm<0, 1, 0><<<dim3(cdiv(3 * C, 64), cdiv(tokens, 128)), 256, 0, stream>>>(
          a16, w16, p.qkv_b, nullptr, qkv16, tokens, 3 * C, C);

      // fused attention -> att16 f16
      k_attn<<<dim3(nWB, nh), 128, 0, stream>>>(
          qkv16, p.rpb, att16, nws, nh, C, H, shift);

      // proj GEMM -> tmp32 fp32 (window layout)
      k_cvt<<<cdiv(C * C, TPB), TPB, 0, stream>>>(p.proj_w, w16, C * C);
      k_gemm<0, 0, 0><<<dim3(cdiv(C, 64), cdiv(tokens, 128)), 256, 0, stream>>>(
          att16, w16, p.proj_b, nullptr, tmp32, tokens, C, C);

      // window reverse + roll back + residual into x
      k_wrev_add<<<cdiv((int)((size_t)tokens * C), TPB), TPB, 0, stream>>>(
          xbuf, tmp32, Bn, H, C, shift);

      // LN2 -> a16 f16
      k_ln<1><<<cdiv(tokens, TPB), TPB, 0, stream>>>(
          xbuf, p.n2g, p.n2b, a16, tokens, C);

      // fc1 GEMM + GELU -> h16 f16
      k_cvt<<<cdiv(C * 4 * C, TPB), TPB, 0, stream>>>(p.fc1_w, w16, C * 4 * C);
      k_gemm<1, 1, 0><<<dim3(cdiv(4 * C, 64), cdiv(tokens, 128)), 256, 0, stream>>>(
          a16, w16, p.fc1_b, nullptr, h16, tokens, 4 * C, C);

      // fc2 GEMM + residual(x) -> x fp32
      k_cvt<<<cdiv(4 * C * C, TPB), TPB, 0, stream>>>(p.fc2_w, w16, 4 * C * C);
      k_gemm<0, 0, 1><<<dim3(cdiv(C, 64), cdiv(tokens, 128)), 256, 0, stream>>>(
          h16, w16, p.fc2_b, xbuf, xbuf, tokens, C, 4 * C);
    }

    // emit stage output
    size_t bytes = (size_t)tokens * C * sizeof(float);
    hipMemcpyAsync((char*)d_out + out_off, xbuf, bytes,
                   hipMemcpyDeviceToDevice, stream);
    out_off += bytes;

    // downsample
    if (s < 3) {
      int H2 = H / 2;
      int mtok = Bn * H2 * H2;
      k_merge_ln<<<cdiv(mtok, TPB), TPB, 0, stream>>>(
          xbuf, ds_ng[s], ds_nb[s], a16, Bn, H, C);
      k_cvt<<<cdiv(4 * C * 2 * C, TPB), TPB, 0, stream>>>(ds_rw[s], w16,
                                                          4 * C * 2 * C);
      k_gemm<0, 0, 0><<<dim3(cdiv(2 * C, 64), cdiv(mtok, 128)), 256, 0, stream>>>(
          a16, w16, nullptr, nullptr, xbuf, mtok, 2 * C, 4 * C);
      H = H2;
      C *= 2;
    }
  }
}
